// Block_88295937671453
// MI455X (gfx1250) — matrix operand order repaired and checked by reading, not compiled, not run on hardware
//
#include <hip/hip_runtime.h>
#include <hip/hip_bf16.h>

// ---------------------------------------------------------------------------
// Transformer block for MI455X (gfx1250, wave32, WMMA).
// All matmuls run as bf16 WMMA (v_wmma_f32_16x16x32_bf16) with f32 accumulate.
// Attention is flash-style: S/P never leave LDS.
// Staging uses GLOBAL_LOAD_ASYNC_TO_LDS_B128 (ASYNCcnt) when the toolchain
// exposes the builtins; otherwise falls back to vector loads + ds stores.
// ---------------------------------------------------------------------------

typedef __attribute__((ext_vector_type(16))) __bf16 v16bf;
typedef __attribute__((ext_vector_type(8)))  float  v8f;

#define B_   2
#define T_   2048
#define C_   1024
#define H_   16
#define HS_  64
#define FF_  4096
#define EPS_ 1e-5f
#define SCALE_ 0.03125f   // C**-0.5 = 1/32 (reference scales by n_embd, not head_size)

#if defined(__gfx1250__) && \
    __has_builtin(__builtin_amdgcn_global_load_async_to_lds_b128) && \
    __has_builtin(__builtin_amdgcn_s_wait_asynccnt)
#define USE_ASYNC_LDS 1
#else
#define USE_ASYNC_LDS 0
#endif

static __device__ __forceinline__ v8f v8f_zero() {
    v8f z = {0.f, 0.f, 0.f, 0.f, 0.f, 0.f, 0.f, 0.f};
    return z;
}

// Copy 16 bf16 (32B) from global to LDS: async (ASYNCcnt) if available.
#if USE_ASYNC_LDS
typedef __attribute__((ext_vector_type(4))) int async_i4;
typedef __attribute__((address_space(1))) async_i4* async_gp;
typedef __attribute__((address_space(3))) async_i4* async_lp;
#endif

static __device__ __forceinline__ void stage16(const __bf16* __restrict__ g,
                                               __bf16* __restrict__ s) {
#if USE_ASYNC_LDS
    __builtin_amdgcn_global_load_async_to_lds_b128((async_gp)(g),     (async_lp)(s),     0, 0);
    __builtin_amdgcn_global_load_async_to_lds_b128((async_gp)(g + 8), (async_lp)(s + 8), 0, 0);
#else
    *reinterpret_cast<uint4*>(s)     = *reinterpret_cast<const uint4*>(g);
    *reinterpret_cast<uint4*>(s + 8) = *reinterpret_cast<const uint4*>(g + 8);
#endif
}

static __device__ __forceinline__ void stage_wait() {
#if USE_ASYNC_LDS
    __builtin_amdgcn_s_wait_asynccnt(0);
#endif
}

// Load a 16x32 bf16 A-fragment (row-major source, stride in halves, multiple of 8).
// Lane l: M = l%16, g = l/16; elem e -> K = g*16 + e  (16 contiguous halves).
static __device__ __forceinline__ v16bf load_a_frag(const __bf16* p /*&src[(m0+lm)*stride + g*16]*/) {
    union { uint4 u[2]; v16bf v; } c;
    c.u[0] = *reinterpret_cast<const uint4*>(p);
    c.u[1] = *reinterpret_cast<const uint4*>(p + 8);
    return c.v;
}

// Load a 32x16 bf16 B-fragment from an N-major (transposed) buffer.
// Lane l: N = l%16, g = l/16; elem e -> K = g*16 + e  (16 contiguous halves).
static __device__ __forceinline__ v16bf load_bt_frag(const __bf16* p /*&srcT[(n0+ln)*stride + g*16]*/) {
    union { uint4 u[2]; v16bf v; } c;
    c.u[0] = *reinterpret_cast<const uint4*>(p);
    c.u[1] = *reinterpret_cast<const uint4*>(p + 8);
    return c.v;
}

// ---------------------------------------------------------------------------
// Generic WMMA GEMM: out = act(A[M,K](bf16) @ B[K,N](bf16) + bias) + resid
// Block tile 128x128, 256 threads (8 waves), wave tile 32x64 (2x4 WMMA tiles).
// Epilogue behavior is compile-time (no per-element branches).
// ---------------------------------------------------------------------------
template<bool BIAS, bool RES, bool RELU, bool OUTF, bool OUTH>
__global__ __launch_bounds__(256)
void gemm_bf16(const __bf16* __restrict__ A, const __bf16* __restrict__ Bm,
               const float* __restrict__ bias, const float* __restrict__ resid,
               float* __restrict__ outF, __bf16* __restrict__ outH,
               int M, int N, int K)
{
    __shared__ __align__(16) __bf16 sA [128 * 40];   // [row][k], stride 40 (16B-aligned rows)
    __shared__ __align__(16) __bf16 sBt[128 * 40];   // [n][k] (transposed tile)

    const int tid  = threadIdx.x;
    const int lane = tid & 31;
    const int wv   = tid >> 5;
    const int mi   = wv >> 1;          // 0..3 : 32-row slab
    const int nj   = wv & 1;           // 0..1 : 64-col slab
    const int lm   = lane & 15;
    const int lg   = lane >> 4;

    const int rowBlk = blockIdx.y * 128;
    const int colBlk = blockIdx.x * 128;

    v8f acc[2][4];
    #pragma unroll
    for (int s = 0; s < 2; ++s)
        #pragma unroll
        for (int t = 0; t < 4; ++t) acc[s][t] = v8f_zero();

    for (int k0 = 0; k0 < K; k0 += 32) {
        __syncthreads();
        {   // stage A tile 128x32: 16 halves per thread (async -> LDS)
            const int r = tid >> 1, off = (tid & 1) * 16;
            stage16(A + (size_t)(rowBlk + r) * K + k0 + off, &sA[r * 40 + off]);
        }
        {   // stage B tile 32x128, transposed into sBt[n][k]
            const int kr = tid >> 3, nOff = (tid & 7) * 16;
            const __bf16* g = Bm + (size_t)(k0 + kr) * N + colBlk + nOff;
            __bf16 tmp[16];
            *reinterpret_cast<uint4*>(tmp)     = *reinterpret_cast<const uint4*>(g);
            *reinterpret_cast<uint4*>(tmp + 8) = *reinterpret_cast<const uint4*>(g + 8);
            #pragma unroll
            for (int e = 0; e < 16; ++e) sBt[(nOff + e) * 40 + kr] = tmp[e];
        }
        stage_wait();
        __syncthreads();

        v16bf afr[2];
        #pragma unroll
        for (int s = 0; s < 2; ++s)
            afr[s] = load_a_frag(&sA[(mi * 32 + s * 16 + lm) * 40 + lg * 16]);

        #pragma unroll
        for (int t = 0; t < 4; ++t) {
            const v16bf bfr = load_bt_frag(&sBt[(nj * 64 + t * 16 + lm) * 40 + lg * 16]);
            #pragma unroll
            for (int s = 0; s < 2; ++s)
                acc[s][t] = __builtin_amdgcn_wmma_f32_16x16x32_bf16(
                    false, afr[s], false, bfr, (short)0, acc[s][t], false, false);
        }
    }

    // epilogue: compile-time bias / residual / relu, f32 and/or bf16 stores
    #pragma unroll
    for (int s = 0; s < 2; ++s)
        #pragma unroll
        for (int t = 0; t < 4; ++t) {
            const int col  = colBlk + nj * 64 + t * 16 + lm;
            const int rowB = rowBlk + mi * 32 + s * 16 + lg * 8;
            const float bv = BIAS ? bias[col] : 0.0f;
            #pragma unroll
            for (int r = 0; r < 8; ++r) {
                float vv = acc[s][t][r] + bv;
                const size_t o = (size_t)(rowB + r) * N + col;
                if (RES)  vv += resid[o];
                if (RELU) vv = fmaxf(vv, 0.0f);
                if (OUTF) outF[o] = vv;
                if (OUTH) outH[o] = (__bf16)vv;
            }
        }
}

// ---------------------------------------------------------------------------
// Flash attention: one block per (b, h, 64-query tile). q/k/v/att are bf16
// [B,T,C] with head h occupying columns [h*64, h*64+64).
// ---------------------------------------------------------------------------
__global__ __launch_bounds__(256)
void attn_flash(const __bf16* __restrict__ qg, const __bf16* __restrict__ kg,
                const __bf16* __restrict__ vg, __bf16* __restrict__ att)
{
    __shared__ __align__(16) __bf16 sQ [64 * 72];
    __shared__ __align__(16) __bf16 sK [64 * 72];   // [key][hs]  (== B^T layout for QK^T)
    __shared__ __align__(16) __bf16 sVt[64 * 72];   // [hs][key]  (== B^T layout for P@V)
    __shared__ float  sS  [64 * 68];
    __shared__ __align__(16) __bf16 sP [64 * 72];
    __shared__ float  sFac[64];

    const int tid  = threadIdx.x;
    const int lane = tid & 31;
    const int wv   = tid >> 5;
    const int mi   = wv & 3;           // 16-row tile within the 64 queries
    const int nh   = wv >> 2;          // owns n-tiles nh*2, nh*2+1
    const int lm   = lane & 15;
    const int lg   = lane >> 4;
    const int m0   = mi * 16;

    const int qt = blockIdx.x;
    const int h  = blockIdx.y;
    const int b  = blockIdx.z;
    const int qBase = qt * 64;
    const size_t headOff = (size_t)h * HS_;

    {   // stage Q (64x64): 16 halves per thread (async -> LDS)
        const int r = tid >> 2, off = (tid & 3) * 16;
        stage16(qg + ((size_t)b * T_ + qBase + r) * C_ + headOff + off, &sQ[r * 72 + off]);
    }
    stage_wait();
    __syncthreads();

    v16bf qfr[2];
    #pragma unroll
    for (int ks = 0; ks < 2; ++ks)
        qfr[ks] = load_a_frag(&sQ[(m0 + lm) * 72 + ks * 32 + lg * 16]);

    v8f o0 = v8f_zero(), o1 = v8f_zero();
    float mRun = -1e30f, lRun = 0.0f;

    for (int kt = 0; kt <= qt; ++kt) {
        const int kBase = kt * 64;
        __syncthreads();
        {   // stage K (natural, async) and V (transposed, manual)
            const int r = tid >> 2, off = (tid & 3) * 16;
            stage16(kg + ((size_t)b * T_ + kBase + r) * C_ + headOff + off, &sK[r * 72 + off]);
            const __bf16* gv = vg + ((size_t)b * T_ + kBase + r) * C_ + headOff + off;
            __bf16 tmp[16];
            *reinterpret_cast<uint4*>(tmp)     = *reinterpret_cast<const uint4*>(gv);
            *reinterpret_cast<uint4*>(tmp + 8) = *reinterpret_cast<const uint4*>(gv + 8);
            #pragma unroll
            for (int e = 0; e < 16; ++e) sVt[(off + e) * 72 + r] = tmp[e];
        }
        stage_wait();
        __syncthreads();

        // S = Q @ K^T  (this wave's two 16x16 n-tiles) -> sS
        #pragma unroll
        for (int t = 0; t < 2; ++t) {
            const int ng = nh * 2 + t;
            v8f acc = v8f_zero();
            #pragma unroll
            for (int ks = 0; ks < 2; ++ks) {
                const v16bf bfr = load_bt_frag(&sK[(ng * 16 + lm) * 72 + ks * 32 + lg * 16]);
                acc = __builtin_amdgcn_wmma_f32_16x16x32_bf16(
                    false, qfr[ks], false, bfr, (short)0, acc, false, false);
            }
            #pragma unroll
            for (int r = 0; r < 8; ++r)
                sS[(m0 + lg * 8 + r) * 68 + ng * 16 + lm] = acc[r];
        }
        __syncthreads();

        // online softmax: one thread per query row
        if (tid < 64) {
            const int lim = (kt == qt) ? tid : 63;         // causal: kBase+j <= qBase+tid
            const float* srow = &sS[tid * 68];
            float mx = -1e30f;
            for (int j = 0; j <= lim; ++j) mx = fmaxf(mx, srow[j] * SCALE_);
            const float mNew = fmaxf(mRun, mx);
            const float fac  = __expf(mRun - mNew);
            float rs = 0.0f;
            __bf16* prow = &sP[tid * 72];
            for (int j = 0; j < 64; ++j) {
                const float pv = (j <= lim) ? __expf(srow[j] * SCALE_ - mNew) : 0.0f;
                prow[j] = (__bf16)pv;
                rs += pv;
            }
            lRun = lRun * fac + rs;
            mRun = mNew;
            sFac[tid] = fac;
        }
        __syncthreads();

        // rescale O, then O += P @ V
        #pragma unroll
        for (int r = 0; r < 8; ++r) {
            const float f = sFac[m0 + lg * 8 + r];
            o0[r] *= f; o1[r] *= f;
        }
        v16bf pfr[2];
        #pragma unroll
        for (int ks = 0; ks < 2; ++ks)
            pfr[ks] = load_a_frag(&sP[(m0 + lm) * 72 + ks * 32 + lg * 16]);
        #pragma unroll
        for (int t = 0; t < 2; ++t) {
            const int ng = nh * 2 + t;
            #pragma unroll
            for (int ks = 0; ks < 2; ++ks) {
                const v16bf vfr = load_bt_frag(&sVt[(ng * 16 + lm) * 72 + ks * 32 + lg * 16]);
                if (t == 0)
                    o0 = __builtin_amdgcn_wmma_f32_16x16x32_bf16(
                        false, pfr[ks], false, vfr, (short)0, o0, false, false);
                else
                    o1 = __builtin_amdgcn_wmma_f32_16x16x32_bf16(
                        false, pfr[ks], false, vfr, (short)0, o1, false, false);
            }
        }
    }

    __syncthreads();
    if (tid < 64) sFac[tid] = 1.0f / lRun;
    __syncthreads();

    #pragma unroll
    for (int t = 0; t < 2; ++t) {
        const int ng = nh * 2 + t;
        #pragma unroll
        for (int r = 0; r < 8; ++r) {
            const int row = m0 + lg * 8 + r;
            const float ov = ((t == 0) ? o0[r] : o1[r]) * sFac[row];
            att[((size_t)b * T_ + qBase + row) * C_ + headOff + ng * 16 + lm] = (__bf16)ov;
        }
    }
}

// ---------------------------------------------------------------------------
// LayerNorm: one block per row (C = 1024, 256 threads x 4 elems), bf16 output.
// ---------------------------------------------------------------------------
__global__ __launch_bounds__(256)
void layernorm_bf16(const float* __restrict__ x, const float* __restrict__ gamma,
                    const float* __restrict__ beta, __bf16* __restrict__ out)
{
    __shared__ float red [256];
    __shared__ float red2[256];
    const int row = blockIdx.x;
    const int tid = threadIdx.x;
    const float* xr = x + (size_t)row * C_;
    float s = 0.f, s2 = 0.f;
    float vals[4];
    #pragma unroll
    for (int i = 0; i < 4; ++i) {
        const float v = xr[tid + i * 256];
        vals[i] = v; s += v; s2 += v * v;
    }
    red[tid] = s; red2[tid] = s2;
    __syncthreads();
    for (int st = 128; st > 0; st >>= 1) {
        if (tid < st) { red[tid] += red[tid + st]; red2[tid] += red2[tid + st]; }
        __syncthreads();
    }
    const float mean = red[0] * (1.0f / C_);
    const float var  = red2[0] * (1.0f / C_) - mean * mean;
    const float rstd = rsqrtf(var + EPS_);
    __bf16* orow = out + (size_t)row * C_;
    #pragma unroll
    for (int i = 0; i < 4; ++i) {
        const int c = tid + i * 256;
        orow[c] = (__bf16)((vals[i] - mean) * rstd * gamma[c] + beta[c]);
    }
}

// ---------------------------------------------------------------------------
// Weight conversion kernels (f32 -> bf16; QKV additionally [H,C,HS] -> [C,H*HS])
// ---------------------------------------------------------------------------
__global__ void cvt_f32_bf16(const float* __restrict__ in, __bf16* __restrict__ out, int n)
{
    for (int i = blockIdx.x * blockDim.x + threadIdx.x; i < n; i += gridDim.x * blockDim.x)
        out[i] = (__bf16)in[i];
}

__global__ void cvt_qkv_weight(const float* __restrict__ in, __bf16* __restrict__ out)
{
    const int n = C_ * C_;
    for (int i = blockIdx.x * blockDim.x + threadIdx.x; i < n; i += gridDim.x * blockDim.x) {
        const int c = i / C_;
        const int r = i - c * C_;
        const int h = r / HS_;
        const int d = r - h * HS_;
        out[i] = (__bf16)in[((size_t)h * C_ + c) * HS_ + d];
    }
}

// ---------------------------------------------------------------------------
// kernel_launch
// ---------------------------------------------------------------------------
extern "C" void kernel_launch(void* const* d_in, const int* in_sizes, int n_in,
                              void* d_out, int out_size, void* d_ws, size_t ws_size,
                              hipStream_t stream)
{
    (void)in_sizes; (void)n_in; (void)out_size; (void)ws_size;

    const float* x   = (const float*)d_in[0];
    const float* Wq  = (const float*)d_in[1];
    const float* Wk  = (const float*)d_in[2];
    const float* Wv  = (const float*)d_in[3];
    const float* Wo  = (const float*)d_in[4];
    const float* bo  = (const float*)d_in[5];
    const float* W1  = (const float*)d_in[6];
    const float* b1  = (const float*)d_in[7];
    const float* W2  = (const float*)d_in[8];
    const float* b2  = (const float*)d_in[9];
    const float* g1  = (const float*)d_in[10];
    const float* be1 = (const float*)d_in[11];
    const float* g2  = (const float*)d_in[12];
    const float* be2 = (const float*)d_in[13];
    float* out = (float*)d_out;

    char* ws = (char*)d_ws;
    size_t off = 0;
    auto take = [&](size_t bytes) -> void* {
        void* p = (void*)(ws + off);
        off += (bytes + 255) & ~(size_t)255;
        return p;
    };
    const size_t NTC = (size_t)B_ * T_ * C_;
    const size_t NCC = (size_t)C_ * C_;
    const size_t NCF = (size_t)C_ * FF_;

    __bf16* h1   = (__bf16*)take(NTC * 2);
    __bf16* h2   = (__bf16*)take(NTC * 2);
    __bf16* qb   = (__bf16*)take(NTC * 2);
    __bf16* kb   = (__bf16*)take(NTC * 2);
    __bf16* vb   = (__bf16*)take(NTC * 2);
    __bf16* attb = (__bf16*)take(NTC * 2);
    float*  x1   = (float*)take(NTC * 4);
    __bf16* f1   = (__bf16*)take((size_t)B_ * T_ * FF_ * 2);
    __bf16* wqp  = (__bf16*)take(NCC * 2);
    __bf16* wkp  = (__bf16*)take(NCC * 2);
    __bf16* wvp  = (__bf16*)take(NCC * 2);
    __bf16* wop  = (__bf16*)take(NCC * 2);
    __bf16* w1p  = (__bf16*)take(NCF * 2);
    __bf16* w2p  = (__bf16*)take(NCF * 2);

    // --- weight conversions ---
    cvt_qkv_weight<<<1024, 256, 0, stream>>>(Wq, wqp);
    cvt_qkv_weight<<<1024, 256, 0, stream>>>(Wk, wkp);
    cvt_qkv_weight<<<1024, 256, 0, stream>>>(Wv, wvp);
    cvt_f32_bf16 <<<1024, 256, 0, stream>>>(Wo, wop, (int)NCC);
    cvt_f32_bf16 <<<2048, 256, 0, stream>>>(W1, w1p, (int)NCF);
    cvt_f32_bf16 <<<2048, 256, 0, stream>>>(W2, w2p, (int)NCF);

    const int M = B_ * T_;   // 4096

    // --- ln1 -> h1 ---
    layernorm_bf16<<<M, 256, 0, stream>>>(x, g1, be1, h1);

    // --- q, k, v projections (bf16 out only) ---
    gemm_bf16<false, false, false, false, true><<<dim3(C_ / 128, M / 128), 256, 0, stream>>>(
        h1, wqp, nullptr, nullptr, nullptr, qb, M, C_, C_);
    gemm_bf16<false, false, false, false, true><<<dim3(C_ / 128, M / 128), 256, 0, stream>>>(
        h1, wkp, nullptr, nullptr, nullptr, kb, M, C_, C_);
    gemm_bf16<false, false, false, false, true><<<dim3(C_ / 128, M / 128), 256, 0, stream>>>(
        h1, wvp, nullptr, nullptr, nullptr, vb, M, C_, C_);

    // --- causal flash attention ---
    attn_flash<<<dim3(T_ / 64, H_, B_), 256, 0, stream>>>(qb, kb, vb, attb);

    // --- out-proj + bias + residual -> x1 (f32) ---
    gemm_bf16<true, true, false, true, false><<<dim3(C_ / 128, M / 128), 256, 0, stream>>>(
        attb, wop, bo, x, x1, nullptr, M, C_, C_);

    // --- ln2 -> h2 ---
    layernorm_bf16<<<M, 256, 0, stream>>>(x1, g2, be2, h2);

    // --- FFN1: relu(h2 @ W1 + b1) -> f1 (bf16) ---
    gemm_bf16<true, false, true, false, true><<<dim3(FF_ / 128, M / 128), 256, 0, stream>>>(
        h2, w1p, b1, nullptr, nullptr, f1, M, FF_, C_);

    // --- FFN2: f1 @ W2 + b2 + x1 -> out (f32) ---
    gemm_bf16<true, true, false, true, false><<<dim3(C_ / 128, M / 128), 256, 0, stream>>>(
        f1, w2p, b2, x1, out, nullptr, M, C_, FF_);
}
